// Ego_attention_state_action_60189671686475
// MI455X (gfx1250) — matrix operand order repaired and checked by reading, not compiled, not run on hardware
//
#include <hip/hip_runtime.h>
#include <hip/hip_fp16.h>
#include <stdint.h>

typedef __attribute__((ext_vector_type(16))) _Float16 v16h;
typedef __attribute__((ext_vector_type(8)))  _Float16 v8h;
typedef __attribute__((ext_vector_type(4)))  _Float16 v4h;
typedef __attribute__((ext_vector_type(2)))  _Float16 v2h;
typedef __attribute__((ext_vector_type(8)))  float    v8f;
typedef unsigned int u32x4 __attribute__((ext_vector_type(4)));
typedef int          i32x8 __attribute__((ext_vector_type(8)));
typedef int          i32x4 __attribute__((ext_vector_type(4)));

#define F   256
#define NH  4
#define DH  64
#define ETOT 32   // 1 ego + 31 others

// ---- WMMA fragment loaders (CDNA5 16x16x32 f16 layouts, wave32) ----
__device__ __forceinline__ v16h load_a_lds(const _Float16* X, int mbase, int ks, int lane) {
  int m  = mbase + (lane & 15);
  int kb = ks * 32 + ((lane >> 4) << 3);
  const _Float16* p = X + m * F + kb;
  v8h lo = *(const v8h*)p;
  v8h hi = *(const v8h*)(p + 16);
  return __builtin_shufflevector(lo, hi, 0,1,2,3,4,5,6,7,8,9,10,11,12,13,14,15);
}
__device__ __forceinline__ v16h load_b_lds(const _Float16* W, int ks, int lane) {
  int n  = lane & 15;
  int kb = ks * 32 + ((lane >> 4) << 3);
  const _Float16* p = W + n * F + kb;
  v8h lo = *(const v8h*)p;
  v8h hi = *(const v8h*)(p + 16);
  return __builtin_shufflevector(lo, hi, 0,1,2,3,4,5,6,7,8,9,10,11,12,13,14,15);
}
__device__ __forceinline__ v8f wmma_f16(v16h a, v16h b, v8f c) {
  return __builtin_amdgcn_wmma_f32_16x16x32_f16(false, a, false, b, (short)0, c, false, false);
}

// ---- Tensor Data Mover: 1-D copy of `n` f16 elements global -> LDS ----
__device__ __forceinline__ void tdm_load_f16(const _Float16* gsrc, _Float16* lds_dst,
                                             uint32_t n) {
  uint64_t ga = (uint64_t)(uintptr_t)gsrc;
  uint32_t la = (uint32_t)(uintptr_t)lds_dst;   // flat-LDS aperture: offset in [31:0]
  u32x4 g0;
  g0[0] = 1u;                                              // count=1, user descriptor
  g0[1] = la;                                              // lds_addr
  g0[2] = (uint32_t)ga;                                    // global_addr[31:0]
  g0[3] = ((uint32_t)(ga >> 32) & 0x01FFFFFFu) | (2u << 30); // addr[56:32] | type=2
  i32x8 g1;
  g1[0] = (int)(1u << 16);               // data_size = 1 (2 bytes), wg_mask = 0
  g1[1] = (int)(n << 16);                // tensor_dim0[15:0] @ [63:48]
  g1[2] = (int)((n >> 16) | (1u << 16)); // tensor_dim0[31:16] | tensor_dim1 = 1
  g1[3] = (int)((n & 0xFFFFu) << 16);    // tile_dim0 @ [127:112]
  g1[4] = 1;                             // tile_dim1 = 1
  g1[5] = (int)n;                        // tensor_dim0_stride[31:0]
  g1[6] = 0;
  g1[7] = 0;
  i32x4 gz4 = {0, 0, 0, 0};
  i32x8 gz8 = {0, 0, 0, 0, 0, 0, 0, 0};
  __builtin_amdgcn_tensor_load_to_lds(g0, g1, gz4, gz4, gz8, 0);
}

// ---- Kernel 0: convert the six 256x256 fp32 weights to f16 (L2-resident) ----
__global__ __launch_bounds__(256) void wconv_kernel(
    const float* __restrict__ Wq,  const float* __restrict__ Wk,
    const float* __restrict__ Wv,  const float* __restrict__ Wok,
    const float* __restrict__ Wov, const float* __restrict__ Wc,
    _Float16* __restrict__ dst) {
  int idx   = blockIdx.x * blockDim.x + threadIdx.x;
  int total = 6 * F * F;
  for (int i = idx; i < total; i += gridDim.x * blockDim.x) {
    int m = i >> 16;
    int j = i & 65535;
    const float* src = (m == 0) ? Wq : (m == 1) ? Wk : (m == 2) ? Wv
                     : (m == 3) ? Wok : (m == 4) ? Wov : Wc;
    dst[i] = (_Float16)src[j];
  }
}

// ---- Kernel 1: ego QKV.  eqkv[b][0:256]=q, [256:512]=k0, [512:768]=v0.
// Double-buffered TDM weight staging: tile nt+1 streams in behind compute.
__global__ __launch_bounds__(256) void egoqkv_kernel(
    const float* __restrict__ ego, const _Float16* __restrict__ w16,
    float* __restrict__ eqkv) {
  __shared__ __align__(32) _Float16 sX[128 * F];     // 64 KB
  __shared__ __align__(32) _Float16 sW[2][16 * F];   // 16 KB ping-pong
  const int tid  = threadIdx.x;
  const int lane = tid & 31, wave = tid >> 5;
  const int row0 = blockIdx.x * 128;

  if (wave == 0) tdm_load_f16(w16, sW[0], 16 * F);   // prefetch tile 0

  const float4* src4 = (const float4*)(ego + (size_t)row0 * F);
  for (int i = tid; i < 128 * (F / 4); i += 256) {
    float4 v = src4[i];
    v4h h = {(_Float16)v.x, (_Float16)v.y, (_Float16)v.z, (_Float16)v.w};
    *(v4h*)(sX + i * 4) = h;
  }
  __syncthreads();

  // A fragments invariant across all 48 N-tiles: hoist into registers.
  const int mbase = wave * 16;
  v16h afrag[8];
#pragma unroll
  for (int ks = 0; ks < 8; ++ks) afrag[ks] = load_a_lds(sX, mbase, ks, lane);

  for (int nt = 0; nt < 48; ++nt) {
    const _Float16* cur = sW[nt & 1];
    _Float16*       nxt = sW[(nt + 1) & 1];
    __syncthreads();                    // all waves done reading `nxt` (tile nt-1)
    if (wave == 0) {
      if (nt + 1 < 48) {
        tdm_load_f16(w16 + (size_t)(nt + 1) * 16 * F, nxt, 16 * F);
        __builtin_amdgcn_s_wait_tensorcnt(1);   // in-order: `cur` complete
      } else {
        __builtin_amdgcn_s_wait_tensorcnt(0);
      }
    }
    __syncthreads();
    v8f acc = {};
#pragma unroll
    for (int ks = 0; ks < 8; ++ks)
      acc = wmma_f16(afrag[ks], load_b_lds(cur, ks, lane), acc);
    int j     = nt * 16 + (lane & 15);
    int rbase = row0 + mbase + ((lane >> 4) << 3);
#pragma unroll
    for (int r = 0; r < 8; ++r)
      eqkv[(size_t)(rbase + r) * 768 + j] = acc[r];
  }
}

// ---- Kernel 2: fused others-KV GEMM + masked softmax + attn*V. ----
__global__ __launch_bounds__(256) void fused_attn_kernel(
    const float* __restrict__ others, const uint8_t* __restrict__ mask,
    const float* __restrict__ eqkv,
    const _Float16* __restrict__ wok16, const _Float16* __restrict__ wov16,
    float* __restrict__ value, float* __restrict__ attn_out) {
  __shared__ __align__(32) _Float16 sX[128 * F];      // 64 KB
  __shared__ __align__(32) _Float16 sK[128 * F];      // 64 KB
  __shared__ __align__(32) _Float16 sV[128 * F];      // 64 KB
  __shared__ __align__(32) _Float16 sWk[2][16 * F];   // 16 KB ping-pong
  __shared__ __align__(32) _Float16 sWv[2][16 * F];   // 16 KB ping-pong
  __shared__ float sQ[4 * F];                         //  4 KB
  const int tid  = threadIdx.x;
  const int lane = tid & 31, wave = tid >> 5;
  const int b0   = blockIdx.x * 4;

  if (wave == 0) {                      // prefetch tile 0 of both weights
    tdm_load_f16(wok16, sWk[0], 16 * F);
    tdm_load_f16(wov16, sWv[0], 16 * F);
  }

  // phase 0: stage X (f32->f16) and q (f32)
  for (int i = tid; i < 128 * (F / 4); i += 256) {
    int row = i >> 6, c4 = i & 63;
    int g = row >> 5, e = row & 31;
    v4h h = {};
    if (e > 0) {
      const float4* p = (const float4*)(others + ((size_t)(b0 + g) * 31 + (e - 1)) * F);
      float4 v = p[c4];
      h = (v4h){(_Float16)v.x, (_Float16)v.y, (_Float16)v.z, (_Float16)v.w};
    }
    *(v4h*)(sX + (size_t)row * F + c4 * 4) = h;
  }
  for (int i = tid; i < 4 * F; i += 256) {
    int g = i >> 8;
    sQ[i] = eqkv[(size_t)(b0 + g) * 768 + (i & 255)];
  }
  __syncthreads();

  // phase 1: K/V projection GEMM; A fragments hoisted across the N loop.
  const int proj = wave >> 2;       // 0 = K (Wok), 1 = V (Wov)
  const int mb0  = (wave & 3) * 32; // two 16-row M-tiles
  _Float16* outKV = proj ? sV : sK;
  v16h af0[8], af1[8];
#pragma unroll
  for (int ks = 0; ks < 8; ++ks) {
    af0[ks] = load_a_lds(sX, mb0, ks, lane);
    af1[ks] = load_a_lds(sX, mb0 + 16, ks, lane);
  }
  for (int nt = 0; nt < 16; ++nt) {
    int cb = nt & 1, nb = (nt + 1) & 1;
    __syncthreads();                    // all waves done reading buffer `nb`
    if (wave == 0) {
      if (nt + 1 < 16) {
        tdm_load_f16(wok16 + (size_t)(nt + 1) * 16 * F, sWk[nb], 16 * F);
        tdm_load_f16(wov16 + (size_t)(nt + 1) * 16 * F, sWv[nb], 16 * F);
        __builtin_amdgcn_s_wait_tensorcnt(2);   // 2 ops/buffer, in-order
      } else {
        __builtin_amdgcn_s_wait_tensorcnt(0);
      }
    }
    __syncthreads();
    const _Float16* sW = proj ? sWv[cb] : sWk[cb];
    v8f acc0 = {}, acc1 = {};
#pragma unroll
    for (int ks = 0; ks < 8; ++ks) {
      v16h b = load_b_lds(sW, ks, lane);
      acc0 = wmma_f16(af0[ks], b, acc0);
      acc1 = wmma_f16(af1[ks], b, acc1);
    }
    int n  = nt * 16 + (lane & 15);
    int rb = (lane >> 4) << 3;
#pragma unroll
    for (int r = 0; r < 8; ++r) {
      outKV[(size_t)(mb0 + rb + r) * F + n]      = (_Float16)acc0[r];
      outKV[(size_t)(mb0 + 16 + rb + r) * F + n] = (_Float16)acc1[r];
    }
  }
  __syncthreads();
  // phase 1b: entity-0 (ego) K/V rows from eqkv
  for (int i = tid; i < 4 * F; i += 256) {
    int g = i >> 8, k = i & 255;
    sK[(size_t)(g * 32) * F + k] = (_Float16)eqkv[(size_t)(b0 + g) * 768 + 256 + k];
    sV[(size_t)(g * 32) * F + k] = (_Float16)eqkv[(size_t)(b0 + g) * 768 + 512 + k];
  }
  __syncthreads();

  // phase 2: attention.  16 (g,h) pairs over 8 waves; lane = entity.
  for (int gh = wave; gh < 16; gh += 8) {
    int g = gh >> 2, h = gh & 3;
    int b = b0 + g;
    const float*    qh = sQ + g * F + h * DH;
    const _Float16* kr = sK + (size_t)(g * 32 + lane) * F + h * DH;
    float s = 0.f;
    for (int d = 0; d < DH; d += 2) {
      v2h kv = *(const v2h*)(kr + d);
      s += qh[d] * (float)kv[0] + qh[d + 1] * (float)kv[1];
    }
    s *= 0.125f;                               // 1/sqrt(64)
    if (mask[(size_t)b * ETOT + lane]) s = -1e9f;
    float m = s;
    for (int off = 16; off > 0; off >>= 1) m = fmaxf(m, __shfl_xor(m, off, 32));
    float p = __expf(s - m);
    float sum = p;
    for (int off = 16; off > 0; off >>= 1) sum += __shfl_xor(sum, off, 32);
    float at = p / sum;
    attn_out[(size_t)b * (NH * ETOT) + h * ETOT + lane] = at;
    float a0 = 0.f, a1 = 0.f;
    for (int e2 = 0; e2 < 32; ++e2) {
      float av = __shfl(at, e2, 32);
      v2h vv = *(const v2h*)(sV + (size_t)(g * 32 + e2) * F + h * DH + 2 * lane);
      a0 += av * (float)vv[0];
      a1 += av * (float)vv[1];
    }
    value[(size_t)b * F + h * DH + 2 * lane]     = a0;
    value[(size_t)b * F + h * DH + 2 * lane + 1] = a1;
  }
}

// ---- Kernel 3: res = 0.5 * (value @ Wc^T + ego) ----
__global__ __launch_bounds__(256) void final_kernel(
    const float* __restrict__ value, const float* __restrict__ ego,
    const _Float16* __restrict__ wc16, float* __restrict__ res) {
  __shared__ __align__(32) _Float16 sX[128 * F];
  __shared__ __align__(32) _Float16 sW[2][16 * F];
  const int tid  = threadIdx.x;
  const int lane = tid & 31, wave = tid >> 5;
  const int row0 = blockIdx.x * 128;

  if (wave == 0) tdm_load_f16(wc16, sW[0], 16 * F);

  const float4* src4 = (const float4*)(value + (size_t)row0 * F);
  for (int i = tid; i < 128 * (F / 4); i += 256) {
    float4 v = src4[i];
    v4h h = {(_Float16)v.x, (_Float16)v.y, (_Float16)v.z, (_Float16)v.w};
    *(v4h*)(sX + i * 4) = h;
  }
  __syncthreads();

  const int mbase = wave * 16;
  v16h afrag[8];
#pragma unroll
  for (int ks = 0; ks < 8; ++ks) afrag[ks] = load_a_lds(sX, mbase, ks, lane);

  for (int nt = 0; nt < 16; ++nt) {
    const _Float16* cur = sW[nt & 1];
    _Float16*       nxt = sW[(nt + 1) & 1];
    __syncthreads();
    if (wave == 0) {
      if (nt + 1 < 16) {
        tdm_load_f16(wc16 + (size_t)(nt + 1) * 16 * F, nxt, 16 * F);
        __builtin_amdgcn_s_wait_tensorcnt(1);
      } else {
        __builtin_amdgcn_s_wait_tensorcnt(0);
      }
    }
    __syncthreads();
    v8f acc = {};
#pragma unroll
    for (int ks = 0; ks < 8; ++ks)
      acc = wmma_f16(afrag[ks], load_b_lds(cur, ks, lane), acc);
    int j     = nt * 16 + (lane & 15);
    int rbase = row0 + mbase + ((lane >> 4) << 3);
#pragma unroll
    for (int r = 0; r < 8; ++r) {
      size_t row = (size_t)(rbase + r);
      res[row * F + j] = 0.5f * (acc[r] + ego[row * F + j]);
    }
  }
}

extern "C" void kernel_launch(void* const* d_in, const int* in_sizes, int n_in,
                              void* d_out, int out_size, void* d_ws, size_t ws_size,
                              hipStream_t stream) {
  const float*   ego    = (const float*)d_in[0];
  const float*   others = (const float*)d_in[1];
  const uint8_t* mask   = (const uint8_t*)d_in[2];   // jnp bool = 1 byte
  const float*   Wq  = (const float*)d_in[3];
  const float*   Wk  = (const float*)d_in[4];
  const float*   Wv  = (const float*)d_in[5];
  const float*   Wok = (const float*)d_in[6];
  const float*   Wov = (const float*)d_in[7];
  const float*   Wc  = (const float*)d_in[8];
  const int B = in_sizes[0] / F;   // 16384

  // workspace layout: [w16: 6*64K halves][eqkv: B*768 f32][value: B*256 f32]
  _Float16* w16   = (_Float16*)d_ws;
  float*    eqkv  = (float*)((char*)d_ws + (size_t)6 * F * F * 2);
  float*    value = (float*)((char*)d_ws + (size_t)6 * F * F * 2 + (size_t)B * 768 * 4);
  float*    res   = (float*)d_out;
  float*    attn  = res + (size_t)B * F;   // outputs concatenated: res, attn

  wconv_kernel<<<96, 256, 0, stream>>>(Wq, Wk, Wv, Wok, Wov, Wc, w16);
  egoqkv_kernel<<<B / 128, 256, 0, stream>>>(ego, w16, eqkv);
  fused_attn_kernel<<<B / 4, 256, 0, stream>>>(others, mask, eqkv,
      w16 + 3 * F * F, w16 + 4 * F * F, value, attn);
  final_kernel<<<B / 128, 256, 0, stream>>>(value, ego, w16 + 5 * F * F, res);
}
